// TransNovo_59648505807529
// MI455X (gfx1250) — matrix operand order repaired and checked by reading, not compiled, not run on hardware
//
#include <hip/hip_runtime.h>
#include <hip/hip_bf16.h>

// ---------------- model constants ----------------
#define NLAYERS 6
#define NHEADS  8
#define DMODEL  512
#define DHEAD   64
#define DFFN    2048
#define VOCAB   30
#define NB      16
#define SEQX    512
#define SEQY    128

typedef __attribute__((ext_vector_type(16))) _Float16 v16h;
typedef __attribute__((ext_vector_type(8)))  _Float16 v8h;
typedef __attribute__((ext_vector_type(4)))  _Float16 v4h;
typedef __attribute__((ext_vector_type(8)))  float    v8f;
typedef __attribute__((ext_vector_type(4)))  float    v4f;

#define NEGI  (-1e30f)
#define NEGTH (-5e29f)

// half-wave (16-lane) reductions; wave32, xor offsets <16 stay in each group
__device__ __forceinline__ float hmax16(float x) {
#pragma unroll
  for (int o = 8; o > 0; o >>= 1) x = fmaxf(x, __shfl_xor(x, o, 32));
  return x;
}
__device__ __forceinline__ float hsum16(float x) {
#pragma unroll
  for (int o = 8; o > 0; o >>= 1) x += __shfl_xor(x, o, 32);
  return x;
}
// 16 f16 from LDS (p[0..7], p[16..23]) as two aligned b128 loads
__device__ __forceinline__ v16h ld16(const _Float16* p) {
  v8h lo = *(const v8h*)p;
  v8h hi = *(const v8h*)(p + 16);
  return __builtin_shufflevector(lo, hi, 0,1,2,3,4,5,6,7,8,9,10,11,12,13,14,15);
}
__device__ __forceinline__ v4h cvt4(v4f f) {
  v4h h;
  h[0] = (_Float16)f[0]; h[1] = (_Float16)f[1];
  h[2] = (_Float16)f[2]; h[3] = (_Float16)f[3];
  return h;
}

// =====================================================================
// WMMA GEMM:  C[M,N] = act(A[M,K] @ W[K,N] + bias (+ resid))
// A row-major lda=K, W row-major ldw=N, C row-major ldc=N.
// Batched via grid.z: z -> (zb=z/Hn, zh=z%Hn) applied through strides.
// Block 256 thr = 8 waves (2x4). Tile 64(M) x 64(N), K step 32.
// Wave: two v_wmma_f32_16x16x32_f16 per k-step (rows wm*32 .. +31).
// Assumes M%64==0, K%32==0 (true for all call sites); N guarded.
// =====================================================================
__launch_bounds__(256)
__global__ void gemm_wmma_k(const float* __restrict__ A, long long aSB, long long aSH,
                            const float* __restrict__ W, long long wSH,
                            const float* __restrict__ bias,
                            const float* __restrict__ resid,
                            float* __restrict__ C, long long cSB, long long cSH,
                            int M, int N, int K, int Hn, int relu)
{
  __shared__ _Float16 As[64][40];   // [m][k]  row = 80B (16B aligned)
  __shared__ _Float16 Bs[64][40];   // [n][k]  fragment-friendly

  const int tid  = threadIdx.x;
  const int lane = tid & 31;
  const int w    = tid >> 5;
  const int wm   = w >> 2;          // 0..1
  const int wn   = w & 3;           // 0..3
  const int mBase = blockIdx.y * 64;
  const int nBase = blockIdx.x * 64;
  const int z  = blockIdx.z;
  const int zb = z / Hn, zh = z % Hn;
  const size_t offA = (size_t)zb * (size_t)aSB + (size_t)zh * (size_t)aSH;
  const size_t offW = (size_t)zh * (size_t)wSH;
  const size_t offC = (size_t)zb * (size_t)cSB + (size_t)zh * (size_t)cSH;
  const bool fullN = (nBase + 64 <= N);

  const v8f z8 = {0.f,0.f,0.f,0.f,0.f,0.f,0.f,0.f};
  v8f acc0 = z8, acc1 = z8;

  // per-thread tile-load coordinates
  const int jA = tid * 8;                 // A: 64x32 = 2048 elems
  const int rA = jA >> 5, kA = jA & 31;
  const int jB = tid * 8;                 // B: 32x64 = 2048 elems
  const int kB = jB >> 6, nB0 = jB & 63;

  for (int k0 = 0; k0 < K; k0 += 32) {
    // ---- A tile (vectorized, unguarded: M%64==0, K%32==0) ----
    {
      const float* src = A + offA + (size_t)(mBase + rA) * K + (k0 + kA);
      v4f f0 = *(const v4f*)src;
      v4f f1 = *(const v4f*)(src + 4);
      *(v4h*)&As[rA][kA]     = cvt4(f0);
      *(v4h*)&As[rA][kA + 4] = cvt4(f1);
    }
    // ---- B tile -> Bs[n][k] ----
    if (fullN) {
      const float* src = W + offW + (size_t)(k0 + kB) * N + (nBase + nB0);
      v4f f0 = *(const v4f*)src;
      v4f f1 = *(const v4f*)(src + 4);
#pragma unroll
      for (int u = 0; u < 4; ++u) Bs[nB0 + u][kB]     = (_Float16)f0[u];
#pragma unroll
      for (int u = 0; u < 4; ++u) Bs[nB0 + 4 + u][kB] = (_Float16)f1[u];
    } else {
      for (int i = tid; i < 32 * 64; i += 256) {
        int kk = i >> 6, n = i & 63;
        int nn = nBase + n;
        float v = (nn < N) ? W[offW + (size_t)(k0 + kk) * N + nn] : 0.f;
        Bs[n][kk] = (_Float16)v;
      }
    }
    __syncthreads();

    // fragments (ISA wave32 layouts), all via b128 LDS loads
    const int ka0 = (lane >> 4) * 8;
    const int mA  = wm * 32 + (lane & 15);
    v16h af0 = ld16(&As[mA][0]);            // placeholder shape; real gather below
    // A frag: elements 0..7 -> k = ka0+0..7 ; 8..15 -> k = 16+ka0+0..7
    {
      v8h lo = *(const v8h*)&As[mA][ka0];
      v8h hi = *(const v8h*)&As[mA][16 + ka0];
      af0 = __builtin_shufflevector(lo, hi, 0,1,2,3,4,5,6,7,8,9,10,11,12,13,14,15);
    }
    v16h af1;
    {
      v8h lo = *(const v8h*)&As[mA + 16][ka0];
      v8h hi = *(const v8h*)&As[mA + 16][16 + ka0];
      af1 = __builtin_shufflevector(lo, hi, 0,1,2,3,4,5,6,7,8,9,10,11,12,13,14,15);
    }
    const int nF  = wn * 16 + (lane & 15);
    const int kb0 = (lane >> 4) * 8;
    v16h bf = ld16(&Bs[nF][kb0]);

    acc0 = __builtin_amdgcn_wmma_f32_16x16x32_f16(false, af0, false, bf, (short)0, acc0, false, false);
    acc1 = __builtin_amdgcn_wmma_f32_16x16x32_f16(false, af1, false, bf, (short)0, acc1, false, false);
    __syncthreads();
  }

  // epilogue: C/D layout row = base + v + 8*(lane>>4), col = lane&15
  const int ng = nBase + wn * 16 + (lane & 15);
  if (ng < N) {
    float bv = bias ? bias[ng] : 0.f;
#pragma unroll
    for (int t = 0; t < 2; ++t) {
      v8f acc = t ? acc1 : acc0;
#pragma unroll
      for (int v = 0; v < 8; ++v) {
        int mg = mBase + wm * 32 + t * 16 + v + 8 * (lane >> 4);
        if (mg < M) {
          float val = acc[v] + bv;
          if (resid) val += resid[offC + (size_t)mg * N + ng];
          if (relu)  val = fmaxf(val, 0.f);
          C[offC + (size_t)mg * N + ng] = val;
        }
      }
    }
  }
}

// =====================================================================
// Fused flash attention (head dim 64, wave32 WMMA).
// Q,K,V: [B,H,S,64]. O written as [B, SQ, H*64] (ldo = DMODEL).
// grid = (ceil(SQ/128), H, B), block 256 = 8 waves, 16 q-rows per wave.
// K/V tiles (32 keys x 64 dims) staged in LDS once per BLOCK and shared
// by all 8 waves. Per 32 keys: 4 WMMAs QK^T + online softmax + P restage
// through per-wave LDS (s_wait_dscnt 0) + 4 WMMAs PV.
// =====================================================================
__launch_bounds__(256)
__global__ void attn_wmma_k(const float* __restrict__ Q, const float* __restrict__ K,
                            const float* __restrict__ V, float* __restrict__ O,
                            const unsigned char* __restrict__ kmask,
                            int H, int SQ, int SK, int ldo, int causal)
{
  __shared__ _Float16 Ks[32][72];       // [key][dim]  row 144B
  __shared__ _Float16 Vs[64][40];       // [dim][key]  row 80B (transposed)
  __shared__ _Float16 Pst[8][16][40];   // per-wave P staging

  const int tid  = threadIdx.x;
  const int lane = tid & 31;
  const int w    = tid >> 5;
  const int b = blockIdx.z, h = blockIdx.y;
  const int q0 = (blockIdx.x * 8 + w) * 16;
  const bool qvalid = (q0 < SQ);

  const int mq  = lane & 15;
  const int ka0 = (lane >> 4) * 8;
  const int nk  = lane & 15;
  const int eb  = (lane >> 4) * 8;
  const int rbase = q0 + 8 * (lane >> 4);

  // ---- Q fragments via b128 loads, pre-scaled by 1/sqrt(64) ----
  v16h aq0, aq1;
  if (qvalid) {
    const float* Qrow = Q + ((size_t)(b * H + h) * SQ + q0 + mq) * DHEAD;
    v4f a0 = *(const v4f*)(Qrow + ka0);
    v4f a1 = *(const v4f*)(Qrow + ka0 + 4);
    v4f a2 = *(const v4f*)(Qrow + 16 + ka0);
    v4f a3 = *(const v4f*)(Qrow + 16 + ka0 + 4);
    v4f b0 = *(const v4f*)(Qrow + 32 + ka0);
    v4f b1 = *(const v4f*)(Qrow + 32 + ka0 + 4);
    v4f b2 = *(const v4f*)(Qrow + 48 + ka0);
    v4f b3 = *(const v4f*)(Qrow + 48 + ka0 + 4);
#pragma unroll
    for (int e = 0; e < 4; ++e) {
      aq0[e]      = (_Float16)(a0[e] * 0.125f);
      aq0[e + 4]  = (_Float16)(a1[e] * 0.125f);
      aq0[e + 8]  = (_Float16)(a2[e] * 0.125f);
      aq0[e + 12] = (_Float16)(a3[e] * 0.125f);
      aq1[e]      = (_Float16)(b0[e] * 0.125f);
      aq1[e + 4]  = (_Float16)(b1[e] * 0.125f);
      aq1[e + 8]  = (_Float16)(b2[e] * 0.125f);
      aq1[e + 12] = (_Float16)(b3[e] * 0.125f);
    }
  }

  float mi[8], li[8];
  v8f oacc[4];
  const v8f z8 = {0.f,0.f,0.f,0.f,0.f,0.f,0.f,0.f};
#pragma unroll
  for (int v = 0; v < 8; ++v) { mi[v] = NEGI; li[v] = 0.f; }
#pragma unroll
  for (int t = 0; t < 4; ++t) oacc[t] = z8;

  // causal: block never needs keys beyond its last q row
  int SKeff = SK;
  if (causal) {
    int qmax = blockIdx.x * 128 + 127;
    if (qmax > SQ - 1) qmax = SQ - 1;
    int lim = ((qmax + 32) / 32) * 32;
    SKeff = (lim < SK) ? lim : SK;
  }

  // per-thread K/V tile load coords: 32x64 = 2048 elems, 8 per thread
  const int jT  = tid * 8;
  const int keyT = jT >> 6, dT = jT & 63;

  for (int kb = 0; kb < SKeff; kb += 32) {
    // ---- stage K tile [key][dim] ----
    {
      const float* src = K + ((size_t)(b * H + h) * SK + kb + keyT) * DHEAD + dT;
      v4f f0 = *(const v4f*)src;
      v4f f1 = *(const v4f*)(src + 4);
      *(v4h*)&Ks[keyT][dT]     = cvt4(f0);
      *(v4h*)&Ks[keyT][dT + 4] = cvt4(f1);
    }
    // ---- stage V tile transposed [dim][key] ----
    {
      const float* src = V + ((size_t)(b * H + h) * SK + kb + keyT) * DHEAD + dT;
      v4f f0 = *(const v4f*)src;
      v4f f1 = *(const v4f*)(src + 4);
#pragma unroll
      for (int u = 0; u < 4; ++u) Vs[dT + u][keyT]     = (_Float16)f0[u];
#pragma unroll
      for (int u = 0; u < 4; ++u) Vs[dT + 4 + u][keyT] = (_Float16)f1[u];
    }
    __syncthreads();

    const bool skip = (!qvalid) || (causal && kb > q0 + 15);
    if (!skip) {
      // scores: S = Q*K^T (two key sub-tiles, two head-dim halves)
      v16h bk00 = ld16(&Ks[nk][eb]);
      v16h bk01 = ld16(&Ks[nk][32 + eb]);
      v16h bk10 = ld16(&Ks[nk + 16][eb]);
      v16h bk11 = ld16(&Ks[nk + 16][32 + eb]);
      v8f s0 = __builtin_amdgcn_wmma_f32_16x16x32_f16(false, aq0, false, bk00, (short)0, z8, false, false);
      s0     = __builtin_amdgcn_wmma_f32_16x16x32_f16(false, aq1, false, bk01, (short)0, s0, false, false);
      v8f s1 = __builtin_amdgcn_wmma_f32_16x16x32_f16(false, aq0, false, bk10, (short)0, z8, false, false);
      s1     = __builtin_amdgcn_wmma_f32_16x16x32_f16(false, aq1, false, bk11, (short)0, s1, false, false);

      const int c0 = kb + (lane & 15);
      const int c1 = c0 + 16;
      const int km0 = kmask ? (int)kmask[(size_t)b * SK + c0] : 0;
      const int km1 = kmask ? (int)kmask[(size_t)b * SK + c1] : 0;

#pragma unroll
      for (int v = 0; v < 8; ++v) {
        float x0 = s0[v], x1 = s1[v];
        int rq = rbase + v;
        if (causal) { if (c0 > rq) x0 = NEGI; if (c1 > rq) x1 = NEGI; }
        if (km0) x0 = NEGI;
        if (km1) x1 = NEGI;
        float tmax = hmax16(fmaxf(x0, x1));
        float mold = mi[v];
        float mnew = fmaxf(mold, tmax);
        float p0 = (x0 <= NEGTH) ? 0.f : __expf(x0 - mnew);
        float p1 = (x1 <= NEGTH) ? 0.f : __expf(x1 - mnew);
        float alpha = (mold <= NEGTH) ? 0.f : __expf(mold - mnew);
        li[v] = li[v] * alpha + hsum16(p0 + p1);
        mi[v] = mnew;
#pragma unroll
        for (int t = 0; t < 4; ++t) oacc[t][v] *= alpha;
        Pst[w][8 * (lane >> 4) + v][lane & 15]        = (_Float16)p0;
        Pst[w][8 * (lane >> 4) + v][16 + (lane & 15)] = (_Float16)p1;
      }
      // CDNA5 split counter: drain this wave's DS stores before re-reading
      asm volatile("s_wait_dscnt 0" ::: "memory");

      // P as A-fragment (b128 LDS loads)
      v16h ap;
      {
        v8h lo = *(const v8h*)&Pst[w][mq][ka0];
        v8h hi = *(const v8h*)&Pst[w][mq][16 + ka0];
        ap = __builtin_shufflevector(lo, hi, 0,1,2,3,4,5,6,7,8,9,10,11,12,13,14,15);
      }
      // O += P @ V
#pragma unroll
      for (int t = 0; t < 4; ++t) {
        v16h bv = ld16(&Vs[t * 16 + nk][eb]);
        oacc[t] = __builtin_amdgcn_wmma_f32_16x16x32_f16(false, ap, false, bv,
                                                         (short)0, oacc[t], false, false);
      }
    }
    __syncthreads();
  }

  if (qvalid) {
#pragma unroll
    for (int t = 0; t < 4; ++t) {
      int col = h * DHEAD + t * 16 + (lane & 15);
#pragma unroll
      for (int v = 0; v < 8; ++v) {
        int row = q0 + 8 * (lane >> 4) + v;
        O[((size_t)b * SQ + row) * ldo + col] = oacc[t][v] / li[v];
      }
    }
  }
}

// =====================================================================
// Small elementwise / reduction kernels
// =====================================================================
__launch_bounds__(256)
__global__ void layernorm_k(const float* __restrict__ x, const float* __restrict__ g,
                            const float* __restrict__ bta, float* __restrict__ y)
{
  __shared__ float s1[256], s2[256];
  const int tid = threadIdx.x;
  const size_t row = blockIdx.x;
  const float* xr = x + row * DMODEL;
  float a0 = xr[tid], a1 = xr[tid + 256];
  s1[tid] = a0 + a1;
  s2[tid] = a0 * a0 + a1 * a1;
  __syncthreads();
  for (int o = 128; o > 0; o >>= 1) {
    if (tid < o) { s1[tid] += s1[tid + o]; s2[tid] += s2[tid + o]; }
    __syncthreads();
  }
  float mean = s1[0] * (1.f / DMODEL);
  float var  = s2[0] * (1.f / DMODEL) - mean * mean;
  float r = rsqrtf(var + 1e-5f);
  y[row * DMODEL + tid]       = (a0 - mean) * r * g[tid] + bta[tid];
  y[row * DMODEL + tid + 256] = (a1 - mean) * r * g[tid + 256] + bta[tid + 256];
}

__global__ void enc_embed_k(const float* __restrict__ X, const float* __restrict__ spec_w,
                            const float* __restrict__ pos, float* __restrict__ enc, int n)
{
  int i = blockIdx.x * blockDim.x + threadIdx.x;
  if (i >= n) return;
  int d = i % DMODEL;
  int bs = i / DMODEL;
  int s = bs % SEQX;
  enc[i] = X[2 * bs] * spec_w[d] + X[2 * bs + 1] * spec_w[DMODEL + d]
         + pos[(size_t)s * DMODEL + d];
}

__global__ void prec_k(const int* __restrict__ charge, const float* __restrict__ pmz,
                       const float* __restrict__ prec_w, float* __restrict__ prec, int n)
{
  int i = blockIdx.x * blockDim.x + threadIdx.x;
  if (i >= n) return;
  int d = i % DMODEL, b = i / DMODEL;
  prec[i] = (float)charge[b] * prec_w[d] + pmz[b] * prec_w[DMODEL + d];
}

__global__ void dec_embed_k(const int* __restrict__ Y, const float* __restrict__ tok,
                            const float* __restrict__ pos, const float* __restrict__ prec,
                            float* __restrict__ dec, int n)
{
  int i = blockIdx.x * blockDim.x + threadIdx.x;
  if (i >= n) return;
  int d = i % DMODEL;
  int bt = i / DMODEL;
  int t = bt % SEQY, b = bt / SEQY;
  int id = Y[b * (SEQY + 1) + t];
  dec[i] = tok[(size_t)id * DMODEL + d] + pos[(size_t)t * DMODEL + d]
         + prec[(size_t)b * DMODEL + d];
}

__global__ void xmask_k(const float* __restrict__ X, unsigned char* __restrict__ m, int n)
{
  int i = blockIdx.x * blockDim.x + threadIdx.x;
  if (i >= n) return;
  m[i] = (X[2 * i] == 0.f && X[2 * i + 1] == 0.f) ? 1 : 0;
}

__global__ void ymask_k(const int* __restrict__ Y, unsigned char* __restrict__ m, int n)
{
  int i = blockIdx.x * blockDim.x + threadIdx.x;
  if (i >= n) return;
  int b = i / SEQY, t = i % SEQY;
  m[i] = (Y[b * (SEQY + 1) + t] == 0) ? 1 : 0;
}

// =====================================================================
// Host orchestration
// =====================================================================
static inline void gemm(hipStream_t st,
                        const float* A, long long aSB, long long aSH,
                        const float* W, long long wSH,
                        const float* bias, const float* resid,
                        float* C, long long cSB, long long cSH,
                        int M, int N, int K, int Hn, int nbat, int relu)
{
  dim3 g((N + 63) / 64, (M + 63) / 64, nbat);
  gemm_wmma_k<<<g, dim3(256), 0, st>>>(A, aSB, aSH, W, wSH, bias, resid,
                                       C, cSB, cSH, M, N, K, Hn, relu);
}

extern "C" void kernel_launch(void* const* d_in, const int* in_sizes, int n_in,
                              void* d_out, int out_size, void* d_ws, size_t ws_size,
                              hipStream_t stream)
{
  (void)in_sizes; (void)n_in; (void)out_size; (void)ws_size;
  const float* X      = (const float*)d_in[0];
  const int*   Y      = (const int*)  d_in[1];
  const int*   charge = (const int*)  d_in[2];
  const float* pmz    = (const float*)d_in[3];
  const float* pos    = (const float*)d_in[4];
  const float* spec_w = (const float*)d_in[5];
  const float* tok    = (const float*)d_in[6];
  const float* prec_w = (const float*)d_in[7];
  const float* e_wq   = (const float*)d_in[8];
  const float* e_wk   = (const float*)d_in[9];
  const float* e_wv   = (const float*)d_in[10];
  const float* e_pw   = (const float*)d_in[11];
  const float* e_pb   = (const float*)d_in[12];
  const float* e_ln1g = (const float*)d_in[13];
  const float* e_ln1b = (const float*)d_in[14];
  const float* e_f1   = (const float*)d_in[15];
  const float* e_fb1  = (const float*)d_in[16];
  const float* e_f2   = (const float*)d_in[17];
  const float* e_fb2  = (const float*)d_in[18];
  const float* e_ln2g = (const float*)d_in[19];
  const float* e_ln2b = (const float*)d_in[20];
  const float* d_sq   = (const float*)d_in[21];
  const float* d_sk   = (const float*)d_in[22];
  const float* d_sv   = (const float*)d_in[23];
  const float* d_spw  = (const float*)d_in[24];
  const float* d_spb  = (const float*)d_in[25];
  const float* d_ln1g = (const float*)d_in[26];
  const float* d_ln1b = (const float*)d_in[27];
  const float* d_cq   = (const float*)d_in[28];
  const float* d_ck   = (const float*)d_in[29];
  const float* d_cv   = (const float*)d_in[30];
  const float* d_cpw  = (const float*)d_in[31];
  const float* d_cpb  = (const float*)d_in[32];
  const float* d_ln2g = (const float*)d_in[33];
  const float* d_ln2b = (const float*)d_in[34];
  const float* d_f1   = (const float*)d_in[35];
  const float* d_fb1  = (const float*)d_in[36];
  const float* d_f2   = (const float*)d_in[37];
  const float* d_fb2  = (const float*)d_in[38];
  const float* d_ln3g = (const float*)d_in[39];
  const float* d_ln3b = (const float*)d_in[40];
  const float* ll_w   = (const float*)d_in[41];
  const float* ll_b   = (const float*)d_in[42];
  float* out = (float*)d_out;

  // workspace carve (floats)
  const size_t SZ_ENC = (size_t)NB * SEQX * DMODEL;
  const size_t SZ_DEC = (size_t)NB * SEQY * DMODEL;
  const size_t SZ_QKV = (size_t)NB * NHEADS * SEQX * DHEAD;
  const size_t SZ_FF  = (size_t)NB * SEQX * DFFN;
  float* f = (float*)d_ws;
  float* enc   = f; f += SZ_ENC;
  float* lnb   = f; f += SZ_ENC;
  float* qb    = f; f += SZ_QKV;
  float* kbuf  = f; f += SZ_QKV;
  float* vbuf  = f; f += SZ_QKV;
  float* attnb = f; f += SZ_ENC;
  float* elb   = f; f += SZ_ENC;
  float* ffb   = f; f += SZ_FF;
  float* dec   = f; f += SZ_DEC;
  float* da    = f; f += SZ_DEC;
  float* dln   = f; f += SZ_DEC;
  float* dln2  = f; f += SZ_DEC;
  float* prec  = f; f += (size_t)NB * DMODEL;
  unsigned char* xmask = (unsigned char*)f;
  unsigned char* ymask = xmask + (size_t)NB * SEQX;

  const int BH = NB * NHEADS;
  const long long WQKV = (long long)DMODEL * DHEAD;

  // ---- embeddings & masks ----
  {
    int n = (int)SZ_ENC;
    enc_embed_k<<<(n + 255) / 256, 256, 0, stream>>>(X, spec_w, pos, enc, n);
    xmask_k<<<(NB * SEQX + 255) / 256, 256, 0, stream>>>(X, xmask, NB * SEQX);
    ymask_k<<<(NB * SEQY + 255) / 256, 256, 0, stream>>>(Y, ymask, NB * SEQY);
    prec_k<<<(NB * DMODEL + 255) / 256, 256, 0, stream>>>(charge, pmz, prec_w, prec, NB * DMODEL);
    int nd = (int)SZ_DEC;
    dec_embed_k<<<(nd + 255) / 256, 256, 0, stream>>>(Y, tok, pos, prec, dec, nd);
  }

  // ---- encoder ----
  for (int l = 0; l < NLAYERS; ++l) {
    const size_t lw = (size_t)l * NHEADS * DMODEL * DHEAD;
    layernorm_k<<<NB * SEQX, 256, 0, stream>>>(enc, e_ln1g + l * DMODEL, e_ln1b + l * DMODEL, lnb);
    gemm(stream, lnb, (long long)SEQX * DMODEL, 0, e_wq + lw, WQKV, nullptr, nullptr,
         qb,   (long long)NHEADS * SEQX * DHEAD, (long long)SEQX * DHEAD, SEQX, DHEAD, DMODEL, NHEADS, BH, 0);
    gemm(stream, lnb, (long long)SEQX * DMODEL, 0, e_wk + lw, WQKV, nullptr, nullptr,
         kbuf, (long long)NHEADS * SEQX * DHEAD, (long long)SEQX * DHEAD, SEQX, DHEAD, DMODEL, NHEADS, BH, 0);
    gemm(stream, lnb, (long long)SEQX * DMODEL, 0, e_wv + lw, WQKV, nullptr, nullptr,
         vbuf, (long long)NHEADS * SEQX * DHEAD, (long long)SEQX * DHEAD, SEQX, DHEAD, DMODEL, NHEADS, BH, 0);
    attn_wmma_k<<<dim3((SEQX + 127) / 128, NHEADS, NB), 256, 0, stream>>>(
        qb, kbuf, vbuf, attnb, xmask, NHEADS, SEQX, SEQX, DMODEL, 0);
    gemm(stream, attnb, 0, 0, e_pw + (size_t)l * DMODEL * DMODEL, 0,
         e_pb + l * DMODEL, lnb, enc, 0, 0, NB * SEQX, DMODEL, DMODEL, 1, 1, 0);
    layernorm_k<<<NB * SEQX, 256, 0, stream>>>(enc, e_ln2g + l * DMODEL, e_ln2b + l * DMODEL, lnb);
    gemm(stream, lnb, 0, 0, e_f1 + (size_t)l * DMODEL * DFFN, 0,
         e_fb1 + l * DFFN, nullptr, ffb, 0, 0, NB * SEQX, DFFN, DMODEL, 1, 1, 1);
    gemm(stream, ffb, 0, 0, e_f2 + (size_t)l * DFFN * DMODEL, 0,
         e_fb2 + l * DMODEL, enc, enc, 0, 0, NB * SEQX, DMODEL, DFFN, 1, 1, 0);
  }

  // ---- decoder ----
  for (int l = 0; l < NLAYERS; ++l) {
    const size_t lw = (size_t)l * NHEADS * DMODEL * DHEAD;
    layernorm_k<<<NB * SEQY, 256, 0, stream>>>(dec, d_ln1g + l * DMODEL, d_ln1b + l * DMODEL, dln);
    gemm(stream, dln, (long long)SEQY * DMODEL, 0, d_sq + lw, WQKV, nullptr, nullptr,
         qb,   (long long)NHEADS * SEQY * DHEAD, (long long)SEQY * DHEAD, SEQY, DHEAD, DMODEL, NHEADS, BH, 0);
    gemm(stream, dln, (long long)SEQY * DMODEL, 0, d_sk + lw, WQKV, nullptr, nullptr,
         kbuf, (long long)NHEADS * SEQY * DHEAD, (long long)SEQY * DHEAD, SEQY, DHEAD, DMODEL, NHEADS, BH, 0);
    gemm(stream, dln, (long long)SEQY * DMODEL, 0, d_sv + lw, WQKV, nullptr, nullptr,
         vbuf, (long long)NHEADS * SEQY * DHEAD, (long long)SEQY * DHEAD, SEQY, DHEAD, DMODEL, NHEADS, BH, 0);
    attn_wmma_k<<<dim3((SEQY + 127) / 128, NHEADS, NB), 256, 0, stream>>>(
        qb, kbuf, vbuf, attnb, ymask, NHEADS, SEQY, SEQY, DMODEL, 1);
    gemm(stream, attnb, 0, 0, d_spw + (size_t)l * DMODEL * DMODEL, 0,
         d_spb + l * DMODEL, dln, da, 0, 0, NB * SEQY, DMODEL, DMODEL, 1, 1, 0);
    layernorm_k<<<NB * SEQX, 256, 0, stream>>>(enc, d_ln2g + l * DMODEL, d_ln2b + l * DMODEL, elb);
    layernorm_k<<<NB * SEQY, 256, 0, stream>>>(da,  d_ln2g + l * DMODEL, d_ln2b + l * DMODEL, dln2);
    gemm(stream, dln2, (long long)SEQY * DMODEL, 0, d_cq + lw, WQKV, nullptr, nullptr,
         qb,   (long long)NHEADS * SEQY * DHEAD, (long long)SEQY * DHEAD, SEQY, DHEAD, DMODEL, NHEADS, BH, 0);
    gemm(stream, elb, (long long)SEQX * DMODEL, 0, d_ck + lw, WQKV, nullptr, nullptr,
         kbuf, (long long)NHEADS * SEQX * DHEAD, (long long)SEQX * DHEAD, SEQX, DHEAD, DMODEL, NHEADS, BH, 0);
    gemm(stream, elb, (long long)SEQX * DMODEL, 0, d_cv + lw, WQKV, nullptr, nullptr,
         vbuf, (long long)NHEADS * SEQX * DHEAD, (long long)SEQX * DHEAD, SEQX, DHEAD, DMODEL, NHEADS, BH, 0);
    attn_wmma_k<<<dim3((SEQY + 127) / 128, NHEADS, NB), 256, 0, stream>>>(
        qb, kbuf, vbuf, attnb, nullptr, NHEADS, SEQY, SEQX, DMODEL, 0);
    gemm(stream, attnb, 0, 0, d_cpw + (size_t)l * DMODEL * DMODEL, 0,
         d_cpb + l * DMODEL, dln2, dec, 0, 0, NB * SEQY, DMODEL, DMODEL, 1, 1, 0);
    layernorm_k<<<NB * SEQY, 256, 0, stream>>>(dec, d_ln3g + l * DMODEL, d_ln3b + l * DMODEL, dln);
    gemm(stream, dln, 0, 0, d_f1 + (size_t)l * DMODEL * DFFN, 0,
         d_fb1 + l * DFFN, nullptr, ffb, 0, 0, NB * SEQY, DFFN, DMODEL, 1, 1, 1);
    gemm(stream, ffb, 0, 0, d_f2 + (size_t)l * DFFN * DMODEL, 0,
         d_fb2 + l * DMODEL, dec, dec, 0, 0, NB * SEQY, DMODEL, DFFN, 1, 1, 0);
  }

  // ---- logits ----
  gemm(stream, dec, 0, 0, ll_w, 0, ll_b, nullptr, out, 0, 0,
       NB * SEQY, VOCAB, DMODEL, 1, 1, 0);
}
